// GCN_43980465111672
// MI455X (gfx1250) — matrix operand order repaired and checked by reading, not compiled, not run on hardware
//
#include <hip/hip_runtime.h>
#include <hip/hip_bf16.h>

#define GCN_N 50000
#define GCN_E 500000
#define DIM 128

typedef __attribute__((ext_vector_type(16))) __bf16 v16bf;
typedef __attribute__((ext_vector_type(8)))  float  v8f;

// ---------------------------------------------------------------------------
// Degree accumulation: deg[dst] += 1 per edge (float atomics, L2-resident)
// ---------------------------------------------------------------------------
__global__ void gcn_deg_accum(const long long* __restrict__ ei, float* __restrict__ deg, int E) {
    int e = blockIdx.x * blockDim.x + threadIdx.x;
    if (e >= E) return;
    int d = (int)ei[(size_t)E + e];
    unsafeAtomicAdd(&deg[d], 1.0f);
}

// deg -> dinv = rsqrt(deg+1), invdeg = 1/(deg+1)   (in-place on deg buffer)
__global__ void gcn_deg_finalize(float* __restrict__ deg, float* __restrict__ invdeg, int n) {
    int i = blockIdx.x * blockDim.x + threadIdx.x;
    if (i >= n) return;
    float d = deg[i] + 1.0f;
    deg[i]    = rsqrtf(d);
    invdeg[i] = 1.0f / d;
}

// ---------------------------------------------------------------------------
// GEMM: C[M x 128] = A[M x 128] * W[128 x 128], fp32 in/out,
// computed as bf16x3 split-precision WMMA (hi*hi + hi*lo + lo*hi), f32 acc.
// W is split hi/lo ONCE per block into transposed LDS arrays, so the steady
// state is pure ds_load_b128 + v_wmma.
// Block = 256 threads = 8 waves; wave w -> rows [blk*128 + 16w, +16).
// Requires M % 16 == 0 (true: 50000 = 3125*16).
// ---------------------------------------------------------------------------
__global__ __launch_bounds__(256)
void gcn_gemm128_bf16x3(const float* __restrict__ A, const float* __restrict__ W,
                        float* __restrict__ C, int M) {
    __shared__ __bf16 Wh[DIM * DIM];   // transposed: Wh[n][k], 32 KB
    __shared__ __bf16 Wl[DIM * DIM];   // transposed: Wl[n][k], 32 KB

    int tid = threadIdx.x;
    for (int i = tid; i < DIM * DIM; i += 256) {
        int k = i >> 7, n = i & 127;
        float f = W[i];
        __bf16 h = (__bf16)f;
        Wh[n * DIM + k] = h;
        Wl[n * DIM + k] = (__bf16)(f - (float)h);
    }
    __syncthreads();

    int waveId = tid >> 5;
    int lane   = tid & 31;
    int rowBase = blockIdx.x * 128 + waveId * 16;
    if (rowBase >= M) return;                 // uniform per-wave exit

    // --- A fragments (16x32 bf16 per K-chunk), contiguous k order (same as B):
    // lanes 0-15: M=lane, K = kb..kb+15, kb = (lane>>4)*16
    int m  = lane & 15;
    int kb = (lane >> 4) * 16;
    const float* xr = A + (size_t)(rowBase + m) * DIM;

    v16bf ah[4], al[4];
    #pragma unroll
    for (int kc = 0; kc < 4; ++kc) {
        #pragma unroll
        for (int j = 0; j < 8; ++j) {
            float f0 = xr[kc * 32 + kb + j];
            float f1 = xr[kc * 32 + kb + 8 + j];
            __bf16 h0 = (__bf16)f0;
            __bf16 h1 = (__bf16)f1;
            ah[kc][j]     = h0;  al[kc][j]     = (__bf16)(f0 - (float)h0);
            ah[kc][8 + j] = h1;  al[kc][8 + j] = (__bf16)(f1 - (float)h1);
        }
    }

    // --- B fragments from LDS (32x16 bf16): lane holds column n = lane&15,
    // K = kbB..kbB+15 with kbB = (lane>>4)*16 ; contiguous in transposed W.
    int nIn = lane & 15;
    int kbB = (lane >> 4) * 16;
    int mSt = (lane >> 4) * 8;                // C layout: VGPR r -> M = r + mSt

    for (int t = 0; t < 8; ++t) {
        v8f acc = {0.f, 0.f, 0.f, 0.f, 0.f, 0.f, 0.f, 0.f};
        int n = t * 16 + nIn;
        #pragma unroll
        for (int kc = 0; kc < 4; ++kc) {
            v16bf bh = *(const v16bf*)&Wh[n * DIM + kc * 32 + kbB];
            v16bf bl = *(const v16bf*)&Wl[n * DIM + kc * 32 + kbB];
            acc = __builtin_amdgcn_wmma_f32_16x16x32_bf16(false, ah[kc], false, bh,
                                                          (short)0, acc, false, false);
            acc = __builtin_amdgcn_wmma_f32_16x16x32_bf16(false, ah[kc], false, bl,
                                                          (short)0, acc, false, false);
            acc = __builtin_amdgcn_wmma_f32_16x16x32_bf16(false, al[kc], false, bh,
                                                          (short)0, acc, false, false);
        }
        #pragma unroll
        for (int r = 0; r < 8; ++r) {
            int rr = rowBase + mSt + r;
            C[(size_t)rr * DIM + t * 16 + nIn] = acc[r];
        }
    }
}

// ---------------------------------------------------------------------------
// Edge scatter: one wave per edge; lane owns 4 channels.
// agg[dst] += h[src] * dinv[src] * dinv[dst]
// ---------------------------------------------------------------------------
__global__ __launch_bounds__(256)
void gcn_scatter(const float* __restrict__ h, const long long* __restrict__ ei,
                 const float* __restrict__ dinv, float* __restrict__ agg, int E) {
    int wave = (blockIdx.x * blockDim.x + threadIdx.x) >> 5;
    int lane = threadIdx.x & 31;
    if (wave >= E) return;
    long long s = ei[wave];
    long long d = ei[(size_t)E + wave];
    float norm = dinv[s] * dinv[d];
    float4 v = ((const float4*)(h + (size_t)s * DIM))[lane];
    float* ap = agg + (size_t)d * DIM + lane * 4;
    unsafeAtomicAdd(ap + 0, v.x * norm);
    unsafeAtomicAdd(ap + 1, v.y * norm);
    unsafeAtomicAdd(ap + 2, v.z * norm);
    unsafeAtomicAdd(ap + 3, v.w * norm);
}

// h1 = relu(agg + h0 * invdeg[row] + b[col])
__global__ void gcn_combine1(const float* __restrict__ agg, const float* __restrict__ h0,
                             const float* __restrict__ invdeg, const float* __restrict__ b,
                             float* __restrict__ h1, int n) {
    int i = blockIdx.x * blockDim.x + threadIdx.x;
    if (i >= n) return;
    int row = i >> 7, col = i & 127;
    float v = agg[i] + h0[i] * invdeg[row] + b[col];
    h1[i] = fmaxf(v, 0.0f);
}

// g = h1 + relu(agg + g * invdeg[row] + b[col])   (residual, in place on g)
__global__ void gcn_combine2(const float* __restrict__ agg, float* __restrict__ g,
                             const float* __restrict__ h1, const float* __restrict__ invdeg,
                             const float* __restrict__ b, int n) {
    int i = blockIdx.x * blockDim.x + threadIdx.x;
    if (i >= n) return;
    int row = i >> 7, col = i & 127;
    float v = agg[i] + g[i] * invdeg[row] + b[col];
    g[i] = h1[i] + fmaxf(v, 0.0f);
}

// out[n, 0:2] = h[n, :] @ Wout + bout
__global__ __launch_bounds__(256)
void gcn_out_proj(const float* __restrict__ h, const float* __restrict__ Wout,
                  const float* __restrict__ bout, float* __restrict__ out, int N) {
    __shared__ float ws[DIM * 2];
    ws[threadIdx.x] = Wout[threadIdx.x];
    __syncthreads();
    int nidx = blockIdx.x * blockDim.x + threadIdx.x;
    if (nidx >= N) return;
    const float* hr = h + (size_t)nidx * DIM;
    float a0 = bout[0], a1 = bout[1];
    #pragma unroll 8
    for (int k = 0; k < DIM; ++k) {
        float v = hr[k];
        a0 += v * ws[k * 2 + 0];
        a1 += v * ws[k * 2 + 1];
    }
    out[nidx * 2 + 0] = a0;
    out[nidx * 2 + 1] = a1;
}

// ---------------------------------------------------------------------------
extern "C" void kernel_launch(void* const* d_in, const int* in_sizes, int n_in,
                              void* d_out, int out_size, void* d_ws, size_t ws_size,
                              hipStream_t stream) {
    const float*     x    = (const float*)d_in[0];
    const long long* ei   = (const long long*)d_in[1];   // int64 edge_index [2, E]
    const float*     W0   = (const float*)d_in[2];
    const float*     b0   = (const float*)d_in[3];
    const float*     W1   = (const float*)d_in[4];
    const float*     b1   = (const float*)d_in[5];
    const float*     Wout = (const float*)d_in[6];
    const float*     bout = (const float*)d_in[7];
    float*           out  = (float*)d_out;

    const int N = GCN_N, E = GCN_E;
    const size_t NPAD = 50176;               // 256B-aligned node-vector stride
    const size_t FEAT = (size_t)N * DIM;     // 6.4M floats

    float* dinv   = (float*)d_ws;            // degrees -> rsqrt(deg+1)
    float* invdeg = dinv + NPAD;             // 1/(deg+1)
    float* hbufA  = invdeg + NPAD;           // h0 / g (N x 128)
    float* aggbuf = hbufA + FEAT;            // aggregation target
    float* h1buf  = aggbuf + FEAT;           // layer-1 activations

    const int T = 256;
    dim3 blkE((E + T - 1) / T);
    dim3 blkN((N + T - 1) / T);
    dim3 blkF(((int)FEAT + T - 1) / T);
    dim3 blkG((N + 127) / 128);
    dim3 blkS((E * 32 + T - 1) / T);

    // degrees + norms
    hipMemsetAsync(dinv, 0, (size_t)N * sizeof(float), stream);
    gcn_deg_accum<<<blkE, T, 0, stream>>>(ei, dinv, E);
    gcn_deg_finalize<<<blkN, T, 0, stream>>>(dinv, invdeg, N);

    // ---- layer 1: h0 = x @ W0 ; aggregate ; relu ----
    hipMemsetAsync(aggbuf, 0, FEAT * sizeof(float), stream);
    gcn_gemm128_bf16x3<<<blkG, T, 0, stream>>>(x, W0, hbufA, N);
    gcn_scatter<<<blkS, T, 0, stream>>>(hbufA, ei, dinv, aggbuf, E);
    gcn_combine1<<<blkF, T, 0, stream>>>(aggbuf, hbufA, invdeg, b0, h1buf, (int)FEAT);

    // ---- layer 2: g = h1 @ W1 ; aggregate ; relu + residual ----
    hipMemsetAsync(aggbuf, 0, FEAT * sizeof(float), stream);
    gcn_gemm128_bf16x3<<<blkG, T, 0, stream>>>(h1buf, W1, hbufA, N);
    gcn_scatter<<<blkS, T, 0, stream>>>(hbufA, ei, dinv, aggbuf, E);
    gcn_combine2<<<blkF, T, 0, stream>>>(aggbuf, hbufA, h1buf, invdeg, b1, (int)FEAT);

    // ---- output projection ----
    gcn_out_proj<<<blkN, T, 0, stream>>>(hbufA, Wout, bout, out, N);
}
